// Mynet_6305011990980
// MI455X (gfx1250) — matrix operand order repaired and checked by reading, not compiled, not run on hardware
//
#include <hip/hip_runtime.h>

// ---------------------------------------------------------------------------
// 3-layer GRU + softmax head, persistent WMMA kernel for gfx1250 (MI455X).
// f16 WMMA (v_wmma_f32_16x16x32_f16) with f32 accumulation; f32 gate math.
// Layer logic peeled so all pointers trace to kernargs -> global_load_b128
// (no flat/generic loads, no DScnt coupling on the WMMA wait path).
// ---------------------------------------------------------------------------

typedef __attribute__((ext_vector_type(16))) _Float16 v16h;
typedef __attribute__((ext_vector_type(8)))  _Float16 v8h;
typedef __attribute__((ext_vector_type(8)))  float    v8f;

constexpr int BATCH  = 64;
constexpr int TSTEPS = 512;
constexpr int INFEAT = 128;
constexpr int HDIM   = 512;
constexpr int NCLS   = 32;
constexpr int NBLK   = 16;   // persistent workgroups
constexpr int NTHR   = 384;  // 12 wave32 per workgroup

__device__ __forceinline__ v8f vzero8() {
  v8f z = {0.f, 0.f, 0.f, 0.f, 0.f, 0.f, 0.f, 0.f};
  return z;
}

// A fragment (16x32 f16, M x K): lane holds row (lane&15); K chunk
// {k0+kh .. +16} contiguous with kh = (lane>>4)*16 (matches B)
__device__ __forceinline__ v16h load_a(const _Float16* A, int stride, int row, int kofs) {
  const _Float16* p = A + (size_t)row * stride + kofs;
  v8h lo = *(const v8h*)p;
  v8h hi = *(const v8h*)(p + 8);
  v16h a;
#pragma unroll
  for (int i = 0; i < 8; ++i) { a[i] = lo[i]; a[8 + i] = hi[i]; }
  return a;
}

// Accumulate A[M x K] * W^T into two 16x16 tiles (rows r0.., r0+16..).
// Wrow already includes (col * K + khB) for this lane.
__device__ __forceinline__ void gemm2(const _Float16* __restrict__ A, int strideA, int K,
                                      const _Float16* __restrict__ Wrow,
                                      int rowA0, int rowA1, int khA,
                                      v8f& c0, v8f& c1) {
  for (int k0 = 0; k0 < K; k0 += 32) {
    v16h b  = *(const v16h*)(Wrow + k0);           // B frag: 16 contiguous f16
    v16h a0 = load_a(A, strideA, rowA0, k0 + khA);
    v16h a1 = load_a(A, strideA, rowA1, k0 + khA);
    c0 = __builtin_amdgcn_wmma_f32_16x16x32_f16(false, a0, false, b, (short)0, c0, false, false);
    c1 = __builtin_amdgcn_wmma_f32_16x16x32_f16(false, a1, false, b, (short)0, c1, false, false);
  }
}

// C/D layout (ISA 7.12.2): element (M = base + (lane>>4)*8 + r, N = lane&15)
__device__ __forceinline__ void stacc(float* s, int gate, int mrow, int ucol, const v8f& c) {
#pragma unroll
  for (int r = 0; r < 8; ++r)
    s[(gate * BATCH + mrow + r) * 32 + ucol] = c[r];
}

__device__ __forceinline__ void grid_barrier(unsigned* cnt, unsigned target) {
  __threadfence();
  __syncthreads();
  if (threadIdx.x == 0) {
    __hip_atomic_fetch_add(cnt, 1u, __ATOMIC_RELEASE, __HIP_MEMORY_SCOPE_AGENT);
    while (__hip_atomic_load(cnt, __ATOMIC_ACQUIRE, __HIP_MEMORY_SCOPE_AGENT) < target)
      __builtin_amdgcn_s_sleep(1);
  }
  __builtin_amdgcn_s_cluster_barrier();  // NOP when not cluster-launched (ClusterID==0)
  __syncthreads();
  __threadfence();
}

struct GruP {
  const _Float16* x16;
  const _Float16* Wih0; const _Float16* Whh0;
  const _Float16* Wih1; const _Float16* Whh1;
  const _Float16* Wih2; const _Float16* Whh2;
  const float* bih0; const float* bhh0;
  const float* bih1; const float* bhh1;
  const float* bih2; const float* bhh2;
  const float* Wout; const float* bout;
  _Float16* h16;   // [parity][layer][BATCH][HDIM]
  float*    h32;   // [parity][layer][BATCH][HDIM]
  float*    out;   // [BATCH][TSTEPS][NCLS]
  unsigned* cnt;
};

// One GRU layer for one WG: two WMMA GEMMs (gi, gh) -> LDS -> gate update.
// All pointer arguments must trace directly to kernargs (global addrspace).
__device__ __forceinline__ void gru_layer(
    int blk, int tid, int gate, int mb, int colW, int rA,
    int khA, int khB, int ucol, int mAcc,
    float* s_gi, float* s_gh,
    const _Float16* __restrict__ Ain, int strideIn, int Kin,
    const _Float16* __restrict__ Wib, const _Float16* __restrict__ Whb,
    const float* __restrict__ bihL, const float* __restrict__ bhhL,
    const _Float16* __restrict__ Ah,            // old h_l (parity p)
    const float* __restrict__ h32rd,            // old h_l f32
    float* __restrict__ h32wr,                  // new h_l f32
    _Float16* __restrict__ h16wr) {             // new h_l f16
  const _Float16* Wi = Wib + (size_t)colW * Kin  + khB;
  const _Float16* Wh = Whb + (size_t)colW * HDIM + khB;

  v8f c0 = vzero8(), c1 = vzero8();
  gemm2(Ain, strideIn, Kin, Wi, mb * 32 + rA, mb * 32 + 16 + rA, khA, c0, c1);
  stacc(s_gi, gate, mAcc,      ucol, c0);
  stacc(s_gi, gate, mAcc + 16, ucol, c1);
  c0 = vzero8(); c1 = vzero8();
  gemm2(Ah, HDIM, HDIM, Wh, mb * 32 + rA, mb * 32 + 16 + rA, khA, c0, c1);
  stacc(s_gh, gate, mAcc,      ucol, c0);
  stacc(s_gh, gate, mAcc + 16, ucol, c1);
  __syncthreads();

  // Gate combine + state update for this WG's 64x32 slab (f32 math).
  for (int idx = tid; idx < BATCH * 32; idx += NTHR) {
    const int m = idx >> 5, u = idx & 31;
    const int jg = blk * 32 + u;
    const float gr = s_gi[(0 * BATCH + m) * 32 + u] + bihL[jg] +
                     s_gh[(0 * BATCH + m) * 32 + u] + bhhL[jg];
    const float gz = s_gi[(1 * BATCH + m) * 32 + u] + bihL[HDIM + jg] +
                     s_gh[(1 * BATCH + m) * 32 + u] + bhhL[HDIM + jg];
    const float gin = s_gi[(2 * BATCH + m) * 32 + u] + bihL[2 * HDIM + jg];
    const float ghn = s_gh[(2 * BATCH + m) * 32 + u] + bhhL[2 * HDIM + jg];
    const float rg = 1.f / (1.f + __expf(-gr));
    const float zg = 1.f / (1.f + __expf(-gz));
    const float ng = tanhf(gin + rg * ghn);
    const float hp = h32rd[(size_t)m * HDIM + jg];
    const float hn = (1.f - zg) * ng + zg * hp;
    h32wr[(size_t)m * HDIM + jg] = hn;
    h16wr[(size_t)m * HDIM + jg] = (_Float16)hn;
  }
}

__global__ __launch_bounds__(NTHR, 1) void k_gru(GruP P) {
  const int tid   = threadIdx.x;
  const int blk   = blockIdx.x;        // 0..15, owns hidden units [32*blk, 32*blk+32)
  const int lane  = tid & 31;
  const int wave  = tid >> 5;          // 0..11
  const int mb    = wave & 1;          // M block: rows [32*mb, 32*mb+32)
  const int jj    = wave >> 1;         // 0..5
  const int gate  = jj >> 1;           // 0:r 1:z 2:n
  const int ucolL = (jj & 1) << 4;     // 0 or 16 within the 32-unit slab
  const int colW  = gate * HDIM + blk * 32 + ucolL + (lane & 15); // W row index
  const int rA    = lane & 15;
  const int khA   = (lane >> 4) << 4;  // 0 / 16
  const int khB   = (lane >> 4) << 4;  // 0 / 16
  const int ucol  = ucolL + (lane & 15);
  const int mAcc  = mb * 32 + ((lane >> 4) << 3);

  __shared__ float s_gi[3 * BATCH * 32];
  __shared__ float s_gh[3 * BATCH * 32];

  constexpr size_t HS = (size_t)BATCH * HDIM;  // one [64][512] state plane

  unsigned gen = 0;
  int p = 0;  // step parity: read h[p], write h[p^1]
  for (int t = 0; t < TSTEPS; ++t) {
    const size_t q0 = (size_t)p * 3, q1 = (size_t)(p ^ 1) * 3;

    // ---- layer 0: input = x_t -------------------------------------------
    gru_layer(blk, tid, gate, mb, colW, rA, khA, khB, ucol, mAcc, s_gi, s_gh,
              P.x16 + (size_t)t * INFEAT, TSTEPS * INFEAT, INFEAT,
              P.Wih0, P.Whh0, P.bih0, P.bhh0,
              P.h16 + (q0 + 0) * HS,
              P.h32 + (q0 + 0) * HS,
              P.h32 + (q1 + 0) * HS,
              P.h16 + (q1 + 0) * HS);
    ++gen; grid_barrier(P.cnt, gen * NBLK);

    // ---- layer 1: input = fresh h1 --------------------------------------
    gru_layer(blk, tid, gate, mb, colW, rA, khA, khB, ucol, mAcc, s_gi, s_gh,
              P.h16 + (q1 + 0) * HS, HDIM, HDIM,
              P.Wih1, P.Whh1, P.bih1, P.bhh1,
              P.h16 + (q0 + 1) * HS,
              P.h32 + (q0 + 1) * HS,
              P.h32 + (q1 + 1) * HS,
              P.h16 + (q1 + 1) * HS);
    ++gen; grid_barrier(P.cnt, gen * NBLK);

    // ---- layer 2: input = fresh h2 --------------------------------------
    gru_layer(blk, tid, gate, mb, colW, rA, khA, khB, ucol, mAcc, s_gi, s_gh,
              P.h16 + (q1 + 1) * HS, HDIM, HDIM,
              P.Wih2, P.Whh2, P.bih2, P.bhh2,
              P.h16 + (q0 + 2) * HS,
              P.h32 + (q0 + 2) * HS,
              P.h32 + (q1 + 2) * HS,
              P.h16 + (q1 + 2) * HS);
    ++gen; grid_barrier(P.cnt, gen * NBLK);

    // ---- output head: WG handles 4 batch rows; one wave per row ---------
    const float* h3 = P.h32 + (q1 + 2) * HS;
    if (tid < 128) {
      const int row = blk * 4 + (tid >> 5);
      const int c = lane;
      const float4* hv = (const float4*)(h3 + (size_t)row * HDIM);
      const float4* wv = (const float4*)(P.Wout + (size_t)c * HDIM);
      float a = P.bout[c];
      for (int k = 0; k < HDIM / 4; ++k) {
        const float4 h4 = hv[k], w4 = wv[k];
        a += h4.x * w4.x + h4.y * w4.y + h4.z * w4.z + h4.w * w4.w;
      }
      float mx = a;
#pragma unroll
      for (int off = 16; off > 0; off >>= 1) mx = fmaxf(mx, __shfl_xor(mx, off, 32));
      const float e = __expf(a - mx);
      float sm = e;
#pragma unroll
      for (int off = 16; off > 0; off >>= 1) sm += __shfl_xor(sm, off, 32);
      P.out[((size_t)row * TSTEPS + t) * NCLS + c] = e / sm;
    }
    p ^= 1;
  }
}

__global__ void k_cvt(const float* __restrict__ s, _Float16* __restrict__ d, int n) {
  const int i = blockIdx.x * blockDim.x + threadIdx.x;
  if (i < n) d[i] = (_Float16)s[i];
}

__global__ void k_init(const float* __restrict__ h1, const float* __restrict__ h2,
                       const float* __restrict__ h3,
                       float* __restrict__ h32, _Float16* __restrict__ h16,
                       unsigned* cnt) {
  const int i = blockIdx.x * blockDim.x + threadIdx.x;
  if (i == 0) *cnt = 0u;
  const int n1 = BATCH * HDIM;
  if (i < 3 * n1) {
    const float* src = (i < n1) ? h1 : ((i < 2 * n1) ? h2 : h3);
    const float v = src[i % n1];
    h32[i] = v;                 // parity-0 region
    h16[i] = (_Float16)v;
  }
}

extern "C" void kernel_launch(void* const* d_in, const int* in_sizes, int n_in,
                              void* d_out, int out_size, void* d_ws, size_t ws_size,
                              hipStream_t stream) {
  (void)in_sizes; (void)n_in; (void)out_size; (void)ws_size;
  const float* x    = (const float*)d_in[0];
  const float* h01  = (const float*)d_in[1];
  const float* h02  = (const float*)d_in[2];
  const float* h03  = (const float*)d_in[3];
  const float* WihF[3] = {(const float*)d_in[4], (const float*)d_in[8],  (const float*)d_in[12]};
  const float* WhhF[3] = {(const float*)d_in[5], (const float*)d_in[9],  (const float*)d_in[13]};
  const float* bihF[3] = {(const float*)d_in[6], (const float*)d_in[10], (const float*)d_in[14]};
  const float* bhhF[3] = {(const float*)d_in[7], (const float*)d_in[11], (const float*)d_in[15]};
  const float* Wout = (const float*)d_in[16];
  const float* bout = (const float*)d_in[17];

  char* ws = (char*)d_ws;
  size_t off = 0;
  auto carve = [&](size_t bytes) -> void* {
    void* p = ws + off;
    off = (off + bytes + 255) & ~(size_t)255;
    return p;
  };
  unsigned* cnt  = (unsigned*)carve(256);
  _Float16* x16  = (_Float16*)carve((size_t)BATCH * TSTEPS * INFEAT * 2);
  const float* wsrc[6] = {WihF[0], WhhF[0], WihF[1], WhhF[1], WihF[2], WhhF[2]};
  const int    wn[6]   = {3 * HDIM * INFEAT, 3 * HDIM * HDIM, 3 * HDIM * HDIM,
                          3 * HDIM * HDIM,   3 * HDIM * HDIM, 3 * HDIM * HDIM};
  _Float16* w16[6];
  for (int i = 0; i < 6; ++i) w16[i] = (_Float16*)carve((size_t)wn[i] * 2);
  float*    h32 = (float*)carve((size_t)2 * 3 * BATCH * HDIM * 4);
  _Float16* h16 = (_Float16*)carve((size_t)2 * 3 * BATCH * HDIM * 2);

  const int nx = BATCH * TSTEPS * INFEAT;
  hipLaunchKernelGGL(k_cvt, dim3((nx + 255) / 256), dim3(256), 0, stream, x, x16, nx);
  for (int i = 0; i < 6; ++i)
    hipLaunchKernelGGL(k_cvt, dim3((wn[i] + 255) / 256), dim3(256), 0, stream,
                       wsrc[i], w16[i], wn[i]);
  hipLaunchKernelGGL(k_init, dim3((3 * BATCH * HDIM + 255) / 256), dim3(256), 0, stream,
                     h01, h02, h03, h32, h16, cnt);

  GruP P{x16,
         w16[0], w16[1], w16[2], w16[3], w16[4], w16[5],
         bihF[0], bhhF[0], bihF[1], bhhF[1], bihF[2], bhhF[2],
         Wout, bout, h16, h32, (float*)d_out, cnt};
  hipLaunchKernelGGL(k_gru, dim3(NBLK), dim3(NTHR), 0, stream, P);
}
